// MPN_72816875537084
// MI455X (gfx1250) — matrix operand order repaired and checked by reading, not compiled, not run on hardware
//
#include <hip/hip_runtime.h>
#include <cstdint>
#include <cstddef>

typedef __attribute__((ext_vector_type(16))) __bf16 v16bf;
typedef __attribute__((ext_vector_type(8)))  float  v8f;

union AFrag { v16bf v; uint4 u[2]; };

__device__ __forceinline__ float bf2f(unsigned int u16) {
  union { unsigned int i; float f; } x;
  x.i = u16 << 16;
  return x.f;
}
__device__ __forceinline__ unsigned short f2bf(float f) {
  union { float f; unsigned int i; } x; x.f = f;
  unsigned int b = x.i;
  b += 0x7FFFu + ((b >> 16) & 1u);   // round-to-nearest-even
  return (unsigned short)(b >> 16);
}
__device__ __forceinline__ unsigned int pack2(float lo, float hi) {
  return (unsigned int)f2bf(lo) | ((unsigned int)f2bf(hi) << 16);
}

// ---------------------------------------------------------------------------
// GEMM: Out[M,HP](bf16) = relu( A@W (+ A2@W2) (+ Cadd) (+ bias) )
// A: bf16 row-major, lda elems, K padded to 32*ksteps with zeros.
// Wp: pre-packed B fragments: ((nt*ksteps + ks)*32 + lane)*16 bf16 each.
// One 16x16 tile per wave, 4 waves per block. grid = (ceil(Mtiles/4), NT).
// ---------------------------------------------------------------------------
template<bool TWO_A, bool ADD_INP, bool BIAS, bool STORE_PRE>
__global__ __launch_bounds__(128)
void gemm_bf16_wmma(const unsigned short* __restrict__ A, int lda,
                    const unsigned short* __restrict__ Wp, int ksteps,
                    const unsigned short* __restrict__ A2, int lda2,
                    const unsigned short* __restrict__ Wp2, int ksteps2,
                    const unsigned short* __restrict__ Cadd,
                    const float* __restrict__ bias,
                    unsigned short* __restrict__ Out,
                    unsigned short* __restrict__ OutPre,
                    int ldo, int Mtiles)
{
  const int lane = threadIdx.x & 31;
  const int wv   = threadIdx.x >> 5;
  const int llow = lane & 15;
  const int hi   = lane >> 4;
  const int mt = blockIdx.x * 4 + wv;
  if (mt >= Mtiles) return;          // wave-uniform: EXEC stays all-ones for WMMA
  const int nt = blockIdx.y;

  v8f c = {0.f, 0.f, 0.f, 0.f, 0.f, 0.f, 0.f, 0.f};

  {
    const unsigned short* ap = A + (size_t)(mt * 16 + llow) * lda + hi * 8;
    const unsigned short* bp = Wp + ((size_t)nt * ksteps * 32 + lane) * 16;
    for (int ks = 0; ks < ksteps; ++ks) {
      AFrag a;
      a.u[0] = *(const uint4*)(ap);        // K = base+0..7
      a.u[1] = *(const uint4*)(ap + 16);   // K = base+16..23
      v16bf b = *(const v16bf*)(bp);
      c = __builtin_amdgcn_wmma_f32_16x16x32_bf16(
              false, a.v, false, b, (short)0, c, false, false);
      ap += 32;
      bp += 512;
    }
  }
  if constexpr (TWO_A) {
    const unsigned short* ap = A2 + (size_t)(mt * 16 + llow) * lda2 + hi * 8;
    const unsigned short* bp = Wp2 + ((size_t)nt * ksteps2 * 32 + lane) * 16;
    for (int ks = 0; ks < ksteps2; ++ks) {
      AFrag a;
      a.u[0] = *(const uint4*)(ap);
      a.u[1] = *(const uint4*)(ap + 16);
      v16bf b = *(const v16bf*)(bp);
      c = __builtin_amdgcn_wmma_f32_16x16x32_bf16(
              false, a.v, false, b, (short)0, c, false, false);
      ap += 32;
      bp += 512;
    }
  }

  // C/D layout: VGPR r, lanes0-15: M=r, N=lane; lanes16-31: M=r+8, N=lane-16
  const int col = nt * 16 + llow;
  size_t ob = (size_t)(mt * 16 + hi * 8) * ldo + col;
  float bv = 0.f;
  if constexpr (BIAS) bv = bias[col];
#pragma unroll
  for (int r = 0; r < 8; ++r) {
    float v = c[r];
    if constexpr (STORE_PRE) OutPre[ob + (size_t)r * ldo] = f2bf(v);
    if constexpr (ADD_INP)   v += bf2f((unsigned int)Cadd[ob + (size_t)r * ldo]);
    v += bv;
    Out[ob + (size_t)r * ldo] = f2bf(v > 0.f ? v : 0.f);
  }
}

// f32 [srcRows,srcCols] -> bf16 [*, dstCols], zero-padded rows/cols
__global__ void cvt_pad_bf16(unsigned short* __restrict__ dst,
                             const float* __restrict__ src,
                             int srcRows, int srcCols, int dstCols, long total)
{
  long t = (long)blockIdx.x * blockDim.x + threadIdx.x;
  if (t >= total) return;
  int  ccol = (int)(t % dstCols);
  long r    = t / dstCols;
  float v = (r < srcRows && ccol < srcCols) ? src[r * (long)srcCols + ccol] : 0.f;
  dst[t] = f2bf(v);
}

// Pack W[row0+krow, col] (f32, ld=srcLd) into per-lane WMMA B fragments (bf16).
// Element j of lane: K = ks*32 + (lane>=16?8:0) + (j<8 ? j : 16+j-8), N = nt*16 + (lane&15).
__global__ void pack_weight_bf16(unsigned short* __restrict__ dst,
                                 const float* __restrict__ src,
                                 int srcLd, int row0, int validRows, int validCols,
                                 int ksteps, int total)
{
  int t = blockIdx.x * blockDim.x + threadIdx.x;
  if (t >= total) return;
  int j    = t & 15;
  int lane = (t >> 4) & 31;
  int blk  = t >> 9;            // nt*ksteps + ks
  int ks   = blk % ksteps;
  int nt   = blk / ksteps;
  int hi = lane >> 4, llow = lane & 15;
  int krow = ks * 32 + hi * 8 + (j < 8 ? j : 16 + (j - 8));
  int col  = nt * 16 + llow;
  float v = (krow < validRows && col < validCols)
              ? src[(size_t)(row0 + krow) * srcLd + col] : 0.f;
  dst[t] = f2bf(v);
}

__global__ void pad_f32(float* __restrict__ dst, const float* __restrict__ src,
                        int n, int total)
{
  int t = blockIdx.x * blockDim.x + threadIdx.x;
  if (t >= total) return;
  dst[t] = (t < n) ? src[t] : 0.f;
}

// amsg[a,:] = sum_k msg[a2b[a,k],:] * w_bonds[a2b[a,k]]   (8 bf16 per thread)
__global__ void aggregate_bf16(unsigned short* __restrict__ amsg,
                               const unsigned short* __restrict__ msg,
                               const int* __restrict__ a2b,
                               const float* __restrict__ w_bonds,
                               int nAtoms, int maxNb, int hp)
{
  int chunks = hp >> 3;
  long t = (long)blockIdx.x * blockDim.x + threadIdx.x;
  if (t >= (long)nAtoms * chunks) return;
  int a  = (int)(t / chunks);
  int c8 = (int)(t % chunks);
  float a0=0,a1=0,a2=0,a3=0,a4=0,a5=0,a6=0,a7=0;
  for (int k = 0; k < maxNb; ++k) {
    int b = a2b[(size_t)a * maxNb + k];
    float w = w_bonds[b];
    uint4 m = *(const uint4*)(msg + (size_t)b * hp + c8 * 8);
    a0 += w * bf2f(m.x & 0xffffu); a1 += w * bf2f(m.x >> 16);
    a2 += w * bf2f(m.y & 0xffffu); a3 += w * bf2f(m.y >> 16);
    a4 += w * bf2f(m.z & 0xffffu); a5 += w * bf2f(m.z >> 16);
    a6 += w * bf2f(m.w & 0xffffu); a7 += w * bf2f(m.w >> 16);
  }
  uint4 o;
  o.x = pack2(a0, a1); o.y = pack2(a2, a3);
  o.z = pack2(a4, a5); o.w = pack2(a6, a7);
  *(uint4*)(amsg + (size_t)a * hp + c8 * 8) = o;
}

// tmpb[b,:] = amsg[b2a[b],:] - msg[b2revb[b],:] * w_bonds[b]
__global__ void bond_update_bf16(unsigned short* __restrict__ tmpb,
                                 const unsigned short* __restrict__ amsg,
                                 const unsigned short* __restrict__ msg,
                                 const int* __restrict__ b2a,
                                 const int* __restrict__ b2revb,
                                 const float* __restrict__ w_bonds,
                                 int nBonds, int hp)
{
  int chunks = hp >> 3;
  long t = (long)blockIdx.x * blockDim.x + threadIdx.x;
  if (t >= (long)nBonds * chunks) return;
  int b  = (int)(t / chunks);
  int c8 = (int)(t % chunks);
  int ia = b2a[b];
  int rb = b2revb[b];
  float w = w_bonds[b];
  uint4 av = *(const uint4*)(amsg + (size_t)ia * hp + c8 * 8);
  uint4 mv = *(const uint4*)(msg  + (size_t)rb * hp + c8 * 8);
  uint4 o;
  o.x = pack2(bf2f(av.x & 0xffffu) - w * bf2f(mv.x & 0xffffu),
              bf2f(av.x >> 16)     - w * bf2f(mv.x >> 16));
  o.y = pack2(bf2f(av.y & 0xffffu) - w * bf2f(mv.y & 0xffffu),
              bf2f(av.y >> 16)     - w * bf2f(mv.y >> 16));
  o.z = pack2(bf2f(av.z & 0xffffu) - w * bf2f(mv.z & 0xffffu),
              bf2f(av.z >> 16)     - w * bf2f(mv.z >> 16));
  o.w = pack2(bf2f(av.w & 0xffffu) - w * bf2f(mv.w & 0xffffu),
              bf2f(av.w >> 16)     - w * bf2f(mv.w >> 16));
  *(uint4*)(tmpb + (size_t)b * hp + c8 * 8) = o;
}

// out[m,col] = deg[m] * sum_i(w*ah) / sum_i(w) over the molecule's 50 atoms
__global__ __launch_bounds__(320)
void mol_readout(float* __restrict__ out,
                 const unsigned short* __restrict__ ah,
                 const float* __restrict__ w_atoms,
                 const float* __restrict__ deg,
                 int atomsPerMol, int h, int hp)
{
  int m = blockIdx.x;
  int col = threadIdx.x;
  int base = 1 + m * atomsPerMol;        // row 0 is the padding atom
  float sum = 0.f, wsum = 0.f;
  for (int i = 0; i < atomsPerMol; ++i) {
    float w = w_atoms[base + i];
    wsum += w;
    if (col < h)
      sum += w * bf2f((unsigned int)ah[(size_t)(base + i) * hp + col]);
  }
  if (col < h)
    out[(size_t)m * h + col] = deg[m] * (sum / wsum);
}

extern "C" void kernel_launch(void* const* d_in, const int* in_sizes, int n_in,
                              void* d_out, int out_size, void* d_ws, size_t ws_size,
                              hipStream_t stream)
{
  const float* f_atoms = (const float*)d_in[0];
  const float* f_bonds = (const float*)d_in[1];
  const float* w_atoms = (const float*)d_in[2];
  const float* w_bonds = (const float*)d_in[3];
  const float* deg     = (const float*)d_in[4];
  const float* W_i     = (const float*)d_in[5];
  const float* W_h     = (const float*)d_in[6];
  const float* W_o     = (const float*)d_in[7];
  const float* b_o     = (const float*)d_in[8];
  const int*   a2b     = (const int*)d_in[9];
  const int*   b2a     = (const int*)d_in[10];
  const int*   b2revb  = (const int*)d_in[11];
  // d_in[12] = mol_index: unused, molecules are contiguous 50-atom blocks
  (void)n_in; (void)out_size; (void)ws_size; (void)b2a; (void)b2revb;

  const int H  = in_sizes[8];              // 300
  const int BOND_FDIM = in_sizes[5] / H;   // 147
  const int NA = in_sizes[2];              // 100001
  const int NB = in_sizes[3];              // 200001
  const int ATOM_FDIM = in_sizes[0] / NA;  // 133
  const int NUM_MOLS = in_sizes[4];        // 2000
  const int APM = (NA - 1) / NUM_MOLS;     // 50
  const int MAXNB = in_sizes[9] / NA;      // 6
  const int DEPTH = 3;

  const int HP = ((H + 31) / 32) * 32;               // 320
  const int NT = HP / 16;                            // 20 N-tiles
  const int KB = ((BOND_FDIM + 31) / 32) * 32;       // 160
  const int KA = ((ATOM_FDIM + 31) / 32) * 32;       // 160
  const int KSb = KB / 32, KSa = KA / 32, KSh = HP / 32;

  const int MTb = (NB + 15) / 16, MPb = MTb * 16;
  const int MTa = (NA + 15) / 16, MPa = MTa * 16;

  uintptr_t wsbase = (uintptr_t)d_ws;
  size_t off = 0;
  auto carve = [&](size_t bytes) -> void* {
    void* p = (void*)(wsbase + off);
    off = (off + bytes + 255) & ~(size_t)255;
    return p;
  };
  unsigned short* fb_bf = (unsigned short*)carve((size_t)MPb * KB * 2);
  unsigned short* fa_bf = (unsigned short*)carve((size_t)MPa * KA * 2);
  unsigned short* inp   = (unsigned short*)carve((size_t)MPb * HP * 2);
  unsigned short* msg   = (unsigned short*)carve((size_t)MPb * HP * 2);
  unsigned short* tmpb  = (unsigned short*)carve((size_t)MPb * HP * 2);
  unsigned short* amsg  = (unsigned short*)carve((size_t)MPa * HP * 2);
  unsigned short* wiP   = (unsigned short*)carve((size_t)NT * KSb * 512 * 2);
  unsigned short* whP   = (unsigned short*)carve((size_t)NT * KSh * 512 * 2);
  unsigned short* woaP  = (unsigned short*)carve((size_t)NT * KSa * 512 * 2);
  unsigned short* womP  = (unsigned short*)carve((size_t)NT * KSh * 512 * 2);
  float*          boP   = (float*)carve((size_t)HP * sizeof(float));
  unsigned short* ah    = tmpb;   // reuse: tmpb is dead after the last W_h GEMM

  const int TB = 256;
  { long tot = (long)MPb * KB;
    cvt_pad_bf16<<<dim3((unsigned)((tot + TB - 1) / TB)), TB, 0, stream>>>(
        fb_bf, f_bonds, NB, BOND_FDIM, KB, tot); }
  { long tot = (long)MPa * KA;
    cvt_pad_bf16<<<dim3((unsigned)((tot + TB - 1) / TB)), TB, 0, stream>>>(
        fa_bf, f_atoms, NA, ATOM_FDIM, KA, tot); }
  { int tot = NT * KSb * 512;
    pack_weight_bf16<<<(tot + TB - 1) / TB, TB, 0, stream>>>(
        wiP, W_i, H, 0, BOND_FDIM, H, KSb, tot); }
  { int tot = NT * KSh * 512;
    pack_weight_bf16<<<(tot + TB - 1) / TB, TB, 0, stream>>>(
        whP, W_h, H, 0, H, H, KSh, tot); }
  { int tot = NT * KSa * 512;
    pack_weight_bf16<<<(tot + TB - 1) / TB, TB, 0, stream>>>(
        woaP, W_o, H, 0, ATOM_FDIM, H, KSa, tot); }
  { int tot = NT * KSh * 512;
    pack_weight_bf16<<<(tot + TB - 1) / TB, TB, 0, stream>>>(
        womP, W_o, H, ATOM_FDIM, H, H, KSh, tot); }
  pad_f32<<<1, HP, 0, stream>>>(boP, b_o, H, HP);

  dim3 gB((MTb + 3) / 4, NT), gA((MTa + 3) / 4, NT), blk(128);

  // inp = f_bonds @ W_i ; msg = relu(inp)
  gemm_bf16_wmma<false, false, false, true><<<gB, blk, 0, stream>>>(
      fb_bf, KB, wiP, KSb, nullptr, 0, nullptr, 0,
      nullptr, nullptr, msg, inp, HP, MTb);

  dim3 gAggA((unsigned)(((long)NA * (HP >> 3) + TB - 1) / TB));
  dim3 gAggB((unsigned)(((long)NB * (HP >> 3) + TB - 1) / TB));
  for (int d = 0; d < DEPTH - 1; ++d) {
    aggregate_bf16<<<gAggA, TB, 0, stream>>>(amsg, msg, a2b, w_bonds, NA, MAXNB, HP);
    bond_update_bf16<<<gAggB, TB, 0, stream>>>(tmpb, amsg, msg,
                                               (const int*)d_in[10], (const int*)d_in[11],
                                               w_bonds, NB, HP);
    // msg = relu(inp + tmpb @ W_h)
    gemm_bf16_wmma<false, true, false, false><<<gB, blk, 0, stream>>>(
        tmpb, HP, whP, KSh, nullptr, 0, nullptr, 0,
        inp, nullptr, msg, nullptr, HP, MTb);
  }
  aggregate_bf16<<<gAggA, TB, 0, stream>>>(amsg, msg, a2b, w_bonds, NA, MAXNB, HP);

  // ah = relu(f_atoms @ W_o[:133] + amsg @ W_o[133:] + b_o)
  gemm_bf16_wmma<true, false, true, false><<<gA, blk, 0, stream>>>(
      fa_bf, KA, woaP, KSa, amsg, HP, womP, KSh,
      nullptr, boP, ah, nullptr, HP, MTa);

  mol_readout<<<NUM_MOLS, HP, 0, stream>>>((float*)d_out, ah, w_atoms, deg, APM, H, HP);
}
